// DeepSeekMoE_61014305407522
// MI455X (gfx1250) — matrix operand order repaired and checked by reading, not compiled, not run on hardware
//
#include <hip/hip_runtime.h>
#include <hip/hip_bf16.h>
#include <cstdint>
#include <cstddef>

// ---------------------------------------------------------------------------
// CDNA5 (gfx1250) MoE: bf16 WMMA GEMMs with f32 accumulation.
// LDS tiles stored K-contiguous with 40-ushort row stride (80B, 16B aligned,
// bank-conflict-free for the documented WMMA 16-bit A/B fragment layouts).
// Staging uses GLOBAL_LOAD_ASYNC_TO_LDS_B128 (ASYNCcnt) when available.
// ---------------------------------------------------------------------------

#define LDS_STRIDE 40   // ushorts per LDS tile row (32 data + 8 pad)

typedef __attribute__((ext_vector_type(16))) __bf16 v16bf;
typedef __attribute__((ext_vector_type(8)))  __bf16 v8bf;
typedef __attribute__((ext_vector_type(8)))  float  v8f;
typedef __attribute__((ext_vector_type(4)))  int    v4i;

// Address-space-qualified pointer types for the async-to-LDS builtin:
//   void __builtin_amdgcn_global_load_async_to_lds_b128(
//       v4i addrspace(1)*, v4i addrspace(3)*, imm int offset, imm int cpol)
// (parameters are NON-const per the compiler diagnostic)
typedef __attribute__((address_space(1))) v4i* gv4i_ptr;
typedef __attribute__((address_space(3))) v4i* lv4i_ptr;

#if defined(__has_builtin)
#if __has_builtin(__builtin_amdgcn_global_load_async_to_lds_b128)
#define HAVE_ASYNC_LDS 1
#endif
#if __has_builtin(__builtin_amdgcn_s_wait_asynccnt)
#define HAVE_WAIT_ASYNCCNT 1
#endif
#endif
#ifndef HAVE_ASYNC_LDS
#define HAVE_ASYNC_LDS 0
#endif
#ifndef HAVE_WAIT_ASYNCCNT
#define HAVE_WAIT_ASYNCCNT 0
#endif

__device__ __forceinline__ void wait_async_lds() {
#if HAVE_ASYNC_LDS
#if HAVE_WAIT_ASYNCCNT
  __builtin_amdgcn_s_wait_asynccnt(0);
#else
  asm volatile("s_wait_asynccnt 0" ::: "memory");
#endif
#endif
}

__device__ __forceinline__ unsigned short f32_to_bf16_rne(float f) {
  union { float f; unsigned u; } v; v.f = f;
  unsigned r = v.u + 0x7FFFu + ((v.u >> 16) & 1u);
  return (unsigned short)(r >> 16);
}

__device__ __forceinline__ float silu_f(float x) {
  return x / (1.0f + __expf(-x));
}

// Copy a 128x32 bf16 tile (K-contiguous rows) from global to LDS.
// 512 chunks of 16B; 256 threads x 2 chunks.
__device__ __forceinline__ void stage_tile(unsigned short* lds,
                                           const unsigned short* g,
                                           int ldg, int tid) {
#pragma unroll
  for (int i = 0; i < 2; ++i) {
    int idx = tid + i * 256;
    int r = idx >> 2, c = idx & 3;
    const unsigned short* src = g + (size_t)r * ldg + c * 8;
    unsigned short* dst = &lds[r * LDS_STRIDE + c * 8];
#if HAVE_ASYNC_LDS
    __builtin_amdgcn_global_load_async_to_lds_b128(
        (gv4i_ptr)src, (lv4i_ptr)dst, 0, 0);
#else
    *(uint4*)dst = *(const uint4*)src;
#endif
  }
}

// A fragment (16x32 bf16): lane holds row M=lane&15; lanes 0-15 carry
// K = 0..7 & 16..23, lanes 16-31 carry K = 8..15 & 24..31 (ISA 7.12.2).
__device__ __forceinline__ v16bf load_fragA(const unsigned short* s,
                                            int rowBase, int lane) {
  int m  = rowBase + (lane & 15);
  int ko = (lane >> 4) << 3;             // 0 or 8
  union { v16bf v; v8bf h[2]; } f;
  f.h[0] = *(const v8bf*)(s + m * LDS_STRIDE + ko);
  f.h[1] = *(const v8bf*)(s + m * LDS_STRIDE + ko + 16);
  return f.v;
}

// B fragment (32x16 bf16), tile stored [N][K]: lane holds col N=lane&15;
// lanes 0-15 carry K = 0..7 & 16..23, lanes 16-31 carry K = 8..15 & 24..31.
__device__ __forceinline__ v16bf load_fragB(const unsigned short* s,
                                            int colBase, int lane) {
  int n  = colBase + (lane & 15);
  int kb = (lane >> 4) << 3;             // 0 or 8
  union { v16bf v; v8bf h[2]; } f;
  f.h[0] = *(const v8bf*)(s + n * LDS_STRIDE + kb);
  f.h[1] = *(const v8bf*)(s + n * LDS_STRIDE + kb + 16);
  return f.v;
}

// MODE 0: outF = A@B                       (f32 store)
// MODE 1: outH = silu(other) * (A@B) * combine[m][eIdx]   (bf16 store)
// MODE 2: outH = silu(A@B + bias[n])                      (bf16 store)
// MODE 3: outF (+)= A@B (+ bias[n])        (accFlag: accumulate)
template <int MODE>
__global__ __launch_bounds__(256)
void moe_gemm_bf16(const unsigned short* __restrict__ A,   // [M][K] bf16
                   const unsigned short* __restrict__ Bt,  // [N][K] bf16
                   int M, int N, int K,
                   float* __restrict__ outF,
                   unsigned short* __restrict__ outH,
                   const float* __restrict__ other,
                   const float* __restrict__ bias,
                   const float* __restrict__ combine,
                   int eIdx, int accFlag) {
  __shared__ __align__(16) unsigned short sA[128 * LDS_STRIDE];
  __shared__ __align__(16) unsigned short sB[128 * LDS_STRIDE];

  const int tid  = threadIdx.x;
  const int lane = tid & 31;
  const int wave = tid >> 5;
  const int wm   = wave >> 1;   // 0..3 -> 32 rows each
  const int wn   = wave & 1;    // 0..1 -> 64 cols each
  const int m0   = blockIdx.y * 128;
  const int n0   = blockIdx.x * 128;

  const v8f vzero = {0.f, 0.f, 0.f, 0.f, 0.f, 0.f, 0.f, 0.f};
  v8f c[2][4];
#pragma unroll
  for (int i = 0; i < 2; ++i)
#pragma unroll
    for (int j = 0; j < 4; ++j) c[i][j] = vzero;

  const int ksteps = K >> 5;
  for (int kk = 0; kk < ksteps; ++kk) {
    stage_tile(sA, A  + (size_t)m0 * K + kk * 32, K, tid);
    stage_tile(sB, Bt + (size_t)n0 * K + kk * 32, K, tid);
    if (kk + 1 < ksteps) {
      // hint next K-block into cache (global_prefetch_b8)
      __builtin_prefetch(A  + (size_t)(m0 + (tid >> 1)) * K + (kk + 1) * 32, 0, 1);
      __builtin_prefetch(Bt + (size_t)(n0 + (tid >> 1)) * K + (kk + 1) * 32, 0, 1);
    }
    wait_async_lds();
    __syncthreads();

    v16bf a[2], b[4];
#pragma unroll
    for (int mt = 0; mt < 2; ++mt) a[mt] = load_fragA(sA, wm * 32 + mt * 16, lane);
#pragma unroll
    for (int nt = 0; nt < 4; ++nt) b[nt] = load_fragB(sB, wn * 64 + nt * 16, lane);

#pragma unroll
    for (int mt = 0; mt < 2; ++mt)
#pragma unroll
      for (int nt = 0; nt < 4; ++nt)
        c[mt][nt] = __builtin_amdgcn_wmma_f32_16x16x32_bf16(
            false, a[mt], false, b[nt], (short)0, c[mt][nt], false, false);

    __syncthreads();
  }

  // Epilogue: C frag VGPR j -> (M = j + 8*(lane>=16), N = lane&15)
#pragma unroll
  for (int mt = 0; mt < 2; ++mt) {
#pragma unroll
    for (int nt = 0; nt < 4; ++nt) {
      const int mBase = m0 + wm * 32 + mt * 16 + ((lane >> 4) << 3);
      const int n     = n0 + wn * 64 + nt * 16 + (lane & 15);
#pragma unroll
      for (int j = 0; j < 8; ++j) {
        const int m = mBase + j;
        const size_t off = (size_t)m * N + n;
        float v = c[mt][nt][j];
        if constexpr (MODE == 0) {
          outF[off] = v;
        } else if constexpr (MODE == 1) {
          float o = other[off];
          float w = combine[m * 8 + eIdx];
          outH[off] = f32_to_bf16_rne(silu_f(o) * v * w);
        } else if constexpr (MODE == 2) {
          outH[off] = f32_to_bf16_rne(silu_f(v + bias[n]));
        } else {
          float t = v;
          if (bias)    t += bias[n];
          if (accFlag) t += outF[off];
          outF[off] = t;
        }
      }
    }
  }
}

// ---------------------------------------------------------------------------
// Router: one wave32 per token. softmax over E=7, top-k on probs+biases,
// renormalized combine weights stored dense [N][8].
// ---------------------------------------------------------------------------
__global__ void moe_router(const float* __restrict__ x,
                           const float* __restrict__ gw,   // [D][7]
                           const float* __restrict__ gb,   // [7]
                           const float* __restrict__ biases,
                           const int* __restrict__ topk_p,
                           float* __restrict__ combine,    // [N][8]
                           int Ntok, int D) {
  const int wave = threadIdx.x >> 5;
  const int lane = threadIdx.x & 31;
  const int n = blockIdx.x * 8 + wave;
  if (n >= Ntok) return;

  float acc[7] = {0.f, 0.f, 0.f, 0.f, 0.f, 0.f, 0.f};
  for (int d = lane; d < D; d += 32) {
    float xv = x[(size_t)n * D + d];
#pragma unroll
    for (int e = 0; e < 7; ++e) acc[e] += xv * gw[d * 7 + e];
  }
#pragma unroll
  for (int s = 16; s > 0; s >>= 1)
#pragma unroll
    for (int e = 0; e < 7; ++e) acc[e] += __shfl_xor(acc[e], s, 32);

  if (lane == 0) {
    float p[7];
    float mx = -1e30f;
#pragma unroll
    for (int e = 0; e < 7; ++e) { p[e] = acc[e] + gb[e]; mx = fmaxf(mx, p[e]); }
    float sum = 0.f;
#pragma unroll
    for (int e = 0; e < 7; ++e) { p[e] = __expf(p[e] - mx); sum += p[e]; }
#pragma unroll
    for (int e = 0; e < 7; ++e) p[e] /= sum;

    int kk = *topk_p; if (kk > 7) kk = 7; if (kk < 1) kk = 1;
    bool used[7] = {false, false, false, false, false, false, false};
    int sel[7];
    float tsum = 0.f;
    for (int t = 0; t < kk; ++t) {
      int best = 0; float bv = -1e30f;
#pragma unroll
      for (int e = 0; e < 7; ++e) {
        float a = p[e] + biases[e];
        if (!used[e] && a > bv) { bv = a; best = e; }
      }
      used[best] = true; sel[t] = best; tsum += p[best];
    }
    float cw[7] = {0.f, 0.f, 0.f, 0.f, 0.f, 0.f, 0.f};
    for (int t = 0; t < kk; ++t) cw[sel[t]] = p[sel[t]] / tsum;
#pragma unroll
    for (int e = 0; e < 7; ++e) combine[n * 8 + e] = cw[e];
    combine[n * 8 + 7] = 0.f;
  }
}

// ---------------------------------------------------------------------------
// Converters
// ---------------------------------------------------------------------------
__global__ void cvt_f32_bf16(const float* __restrict__ src,
                             unsigned short* __restrict__ dst, size_t n) {
  size_t i = (size_t)blockIdx.x * blockDim.x + threadIdx.x;
  size_t stride = (size_t)gridDim.x * blockDim.x;
  for (; i < n; i += stride) dst[i] = f32_to_bf16_rne(src[i]);
}

// src [b][R][C] f32 -> dst [b][C][R] bf16 (tiled transpose-convert)
__global__ void transpose_cvt(const float* __restrict__ src,
                              unsigned short* __restrict__ dst,
                              int R, int C) {
  __shared__ float tile[32][33];
  const int b = blockIdx.z;
  const float* s = src + (size_t)b * R * C;
  unsigned short* d = dst + (size_t)b * R * C;
  const int c0 = blockIdx.x * 32, r0 = blockIdx.y * 32;
  const int tx = threadIdx.x, ty = threadIdx.y;  // 32 x 8
#pragma unroll
  for (int i = 0; i < 4; ++i)
    tile[ty + i * 8][tx] = s[(size_t)(r0 + ty + i * 8) * C + (c0 + tx)];
  __syncthreads();
#pragma unroll
  for (int i = 0; i < 4; ++i)
    d[(size_t)(c0 + ty + i * 8) * R + (r0 + tx)] =
        f32_to_bf16_rne(tile[tx][ty + i * 8]);
}

// ---------------------------------------------------------------------------
// Host orchestration
// ---------------------------------------------------------------------------
extern "C" void kernel_launch(void* const* d_in, const int* in_sizes, int n_in,
                              void* d_out, int out_size, void* d_ws, size_t ws_size,
                              hipStream_t stream) {
  constexpr int Bb = 4, S = 2048, D = 1024, H = 1024, E = 7;
  constexpr int NT = Bb * S;  // 8192 tokens
  (void)in_sizes; (void)n_in; (void)out_size; (void)ws_size;

  const float* x   = (const float*)d_in[0];
  const float* gw  = (const float*)d_in[1];
  const float* gb  = (const float*)d_in[2];
  const float* bia = (const float*)d_in[3];
  const float* w1  = (const float*)d_in[4];
  const float* wg  = (const float*)d_in[5];
  const float* w2  = (const float*)d_in[6];
  const float* sw1 = (const float*)d_in[7];
  const float* sb1 = (const float*)d_in[8];
  const float* sw2 = (const float*)d_in[9];
  const float* sb2 = (const float*)d_in[10];
  const int*   tk  = (const int*)d_in[11];
  float* out = (float*)d_out;

  char* ws = (char*)d_ws;
  size_t off = 0;
  auto wsalloc = [&](size_t bytes) -> void* {
    void* p = ws + off;
    off = (off + bytes + 255) & ~(size_t)255;
    return p;
  };
  unsigned short* xb   = (unsigned short*)wsalloc((size_t)NT * D * 2);
  unsigned short* w1t  = (unsigned short*)wsalloc((size_t)E * D * H * 2);
  unsigned short* wgt  = (unsigned short*)wsalloc((size_t)E * D * H * 2);
  unsigned short* w2t  = (unsigned short*)wsalloc((size_t)E * H * D * 2);
  unsigned short* sw1t = (unsigned short*)wsalloc((size_t)D * H * 2);
  unsigned short* sw2t = (unsigned short*)wsalloc((size_t)H * D * 2);
  float* combine       = (float*)wsalloc((size_t)NT * 8 * 4);
  float* bufC          = (float*)wsalloc((size_t)NT * H * 4);
  unsigned short* h1   = (unsigned short*)wsalloc((size_t)NT * H * 2);

  // 1) One-time precision conversion / weight transpose ([N][K] layouts).
  cvt_f32_bf16<<<4096, 256, 0, stream>>>(x, xb, (size_t)NT * D);
  dim3 tb(32, 8);
  transpose_cvt<<<dim3(H / 32, D / 32, E), tb, 0, stream>>>(w1,  w1t,  D, H);
  transpose_cvt<<<dim3(H / 32, D / 32, E), tb, 0, stream>>>(wg,  wgt,  D, H);
  transpose_cvt<<<dim3(D / 32, H / 32, E), tb, 0, stream>>>(w2,  w2t,  H, D);
  transpose_cvt<<<dim3(H / 32, D / 32, 1), tb, 0, stream>>>(sw1, sw1t, D, H);
  transpose_cvt<<<dim3(D / 32, H / 32, 1), tb, 0, stream>>>(sw2, sw2t, H, D);

  // 2) Router -> dense combine weights [N][8].
  moe_router<<<NT / 8, 256, 0, stream>>>(x, gw, gb, bia, tk, combine, NT, D);

  const dim3 gUp(H / 128, NT / 128);    // (8, 64)
  const dim3 gDown(D / 128, NT / 128);

  // 3) Shared expert: h = silu(x@sw1 + sb1); out = h@sw2 + sb2 (init).
  moe_gemm_bf16<2><<<gUp, 256, 0, stream>>>(xb, sw1t, NT, H, D,
                                            nullptr, h1, nullptr, sb1, nullptr, 0, 0);
  moe_gemm_bf16<3><<<gDown, 256, 0, stream>>>(h1, sw2t, NT, D, H,
                                              out, nullptr, nullptr, sb2, nullptr, 0, 0);

  // 4) Routed experts (dense; combine weight is 0 for unselected experts).
  for (int e = 0; e < E; ++e) {
    const unsigned short* w1e = w1t + (size_t)e * D * H;
    const unsigned short* wge = wgt + (size_t)e * D * H;
    const unsigned short* w2e = w2t + (size_t)e * H * D;
    moe_gemm_bf16<0><<<gUp, 256, 0, stream>>>(xb, w1e, NT, H, D,
                                              bufC, nullptr, nullptr, nullptr, nullptr, 0, 0);
    moe_gemm_bf16<1><<<gUp, 256, 0, stream>>>(xb, wge, NT, H, D,
                                              nullptr, h1, bufC, nullptr, combine, e, 0);
    moe_gemm_bf16<3><<<gDown, 256, 0, stream>>>(h1, w2e, NT, D, H,
                                                out, nullptr, nullptr, nullptr, nullptr, 0, 1);
  }
}
